// ExposedLSTM_80290118632223
// MI455X (gfx1250) — matrix operand order repaired and checked by reading, not compiled, not run on hardware
//
#include <hip/hip_runtime.h>
#include <stdint.h>

// ---------------------------------------------------------------------------
// Types for CDNA5 WMMA (wave32): v16bf A/B fragments, v8f C/D accumulator.
// ---------------------------------------------------------------------------
typedef __bf16 bf16_t;
typedef __attribute__((ext_vector_type(16))) __bf16 v16bf;
typedef __attribute__((ext_vector_type(8)))  float  v8f;

struct alignas(16) U4 { uint32_t x, y, z, w; };

union FragBF {
    U4    q[2];   // two global_load_b128's worth
    v16bf v;      // 16 bf16 values for wmma
};

__device__ __forceinline__ U4 ld128(const bf16_t* p) {
    return *reinterpret_cast<const U4*>(p);
}

__device__ __forceinline__ float sigm(float x) {
    return 1.0f / (1.0f + __expf(-x));
}
__device__ __forceinline__ float tanh_fast(float x) {
    // tanh(x) = 1 - 2/(exp(2x)+1); saturates correctly at +-1 for large |x|
    return 1.0f - 2.0f / (1.0f + __expf(2.0f * x));
}

// ---------------------------------------------------------------------------
// fp32 -> bf16 conversion (grid-stride)
// ---------------------------------------------------------------------------
__global__ void cvt_f32_bf16(const float* __restrict__ s, bf16_t* __restrict__ d,
                             long long n) {
    long long i  = (long long)blockIdx.x * blockDim.x + threadIdx.x;
    long long st = (long long)gridDim.x * blockDim.x;
    for (; i < n; i += st) d[i] = (bf16_t)s[i];
}

// ---------------------------------------------------------------------------
// One LSTM time step (two independent (layer,t) jobs per launch).
//
// Per wave: one 16x16 (batch m x unit u) tile, all 4 gate blocks:
//   acc[g] += x_t[16x512] @ Wih[g-block]^T  (+ h_{t-1}[16x512] @ Whh^T)
// K-loop: 16 iterations; per iteration ALL fragment loads are issued first
// (one clause), then 8 x v_wmma_f32_16x16x32_bf16, so loads overlap WMMA.
// Then gates -> activations -> c/h update -> 6 fp32 outputs + bf16 h feed.
//
// Sizes hardcoded: B=32, H=512, 4H=2048, K=512.
// ---------------------------------------------------------------------------
struct StepArgs {
    const bf16_t* xrow;    // layer input at time t         [32 x 512] bf16
    const bf16_t* hprev;   // h_{t-1} bf16 (null at t==0)   [32 x 512]
    const bf16_t* wih;     // [2048 x 512] bf16
    const bf16_t* whh;     // [2048 x 512] bf16
    const float*  bih;     // [2048]
    const float*  bhh;     // [2048]
    const float*  cprev;   // c_{t-1} fp32 in d_out (null at t==0)
    bf16_t*       hout;    // bf16 h_t for next step / next layer
    unsigned long long outOff;  // ((l*T+t)*B)*H element offset in d_out
};

template <bool HAS_H>
__device__ __forceinline__ void step_body(const StepArgs& A,
                                          float* __restrict__ out,
                                          unsigned long long qs,
                                          int tile, int lane) {
    const int m0  = (tile & 1) << 4;   // batch tile: rows 0-15 / 16-31
    const int u0  = (tile >> 1) << 4;  // unit tile: 32 tiles of 16
    const int r15 = lane & 15;
    const int hi  = lane >> 4;
    const int kbA = hi << 3;           // A frag K offset: 0 or 8
    const int kbB = hi << 3;           // B frag K offset: 0 or 8

    constexpr size_t GSTRIDE = (size_t)512 * 512;  // one gate block of W

    v8f acc[4];
    #pragma unroll
    for (int g = 0; g < 4; ++g) acc[g] = {};

    const bf16_t* arow = A.xrow + (size_t)(m0 + r15) * 512 + kbA;
    const bf16_t* hrow = HAS_H ? (A.hprev + (size_t)(m0 + r15) * 512 + kbA)
                               : arow;
    const bf16_t* wi0  = A.wih + (size_t)(u0 + r15) * 512 + kbB;
    const bf16_t* wh0  = A.whh + (size_t)(u0 + r15) * 512 + kbB;

    #pragma unroll 2
    for (int kk = 0; kk < 512; kk += 32) {
        // ---- issue ALL loads for this K-chunk first (one clause) ----
        FragBF ax, ah, bi[4], bh[4];
        ax.q[0] = ld128(arow + kk);
        ax.q[1] = ld128(arow + kk + 16);
        if (HAS_H) {
            ah.q[0] = ld128(hrow + kk);
            ah.q[1] = ld128(hrow + kk + 16);
        }
        #pragma unroll
        for (int g = 0; g < 4; ++g) {
            bi[g].q[0] = ld128(wi0 + g * GSTRIDE + kk);
            bi[g].q[1] = ld128(wi0 + g * GSTRIDE + kk + 16);
        }
        if (HAS_H) {
            #pragma unroll
            for (int g = 0; g < 4; ++g) {
                bh[g].q[0] = ld128(wh0 + g * GSTRIDE + kk);
                bh[g].q[1] = ld128(wh0 + g * GSTRIDE + kk + 16);
            }
        }
        // ---- then 8 (or 4) back-to-back WMMAs ----
        #pragma unroll
        for (int g = 0; g < 4; ++g)
            acc[g] = __builtin_amdgcn_wmma_f32_16x16x32_bf16(
                false, ax.v, false, bi[g].v, (short)0, acc[g], false, false);
        if (HAS_H) {
            #pragma unroll
            for (int g = 0; g < 4; ++g)
                acc[g] = __builtin_amdgcn_wmma_f32_16x16x32_bf16(
                    false, ah.v, false, bh[g].v, (short)0, acc[g], false, false);
        }
    }

    // C/D layout: reg r, lane -> (m = m0 + (lane>>4)*8 + r, n = u0 + (lane&15))
    const int n = u0 + r15;
    float bsum[4];
    #pragma unroll
    for (int g = 0; g < 4; ++g) bsum[g] = A.bih[g * 512 + n] + A.bhh[g * 512 + n];

    #pragma unroll
    for (int r = 0; r < 8; ++r) {
        const int m = m0 + (hi << 3) + r;
        const float iv = sigm(acc[0][r] + bsum[0]);
        const float fv = sigm(acc[1][r] + bsum[1]);
        const float gv = tanh_fast(acc[2][r] + bsum[2]);
        const float ov = sigm(acc[3][r] + bsum[3]);
        const float cp = HAS_H ? A.cprev[(size_t)m * 512 + n] : 0.0f;
        const float cn = fv * cp + iv * gv;
        const float hn = ov * tanh_fast(cn);
        const unsigned long long idx = A.outOff + (unsigned long long)m * 512 + n;
        out[idx]          = hn;   // h
        out[qs + idx]     = cn;   // c
        out[2 * qs + idx] = iv;   // i
        out[3 * qs + idx] = fv;   // f
        out[4 * qs + idx] = gv;   // g
        out[5 * qs + idx] = ov;   // o
        A.hout[(size_t)m * 512 + n] = (bf16_t)hn;
    }
}

__global__ __launch_bounds__(32)
void lstm_step(StepArgs a0, StepArgs a1, float* __restrict__ out,
               unsigned long long qs) {
    const int job  = blockIdx.x >> 6;             // uniform: which (layer,t)
    const StepArgs A = (job == 0) ? a0 : a1;
    const int tile = blockIdx.x & 63;
    const int lane = threadIdx.x;                 // 0..31 (wave32)
    if (A.hprev != nullptr)                        // uniform: t==0 or not
        step_body<true>(A, out, qs, tile, lane);
    else
        step_body<false>(A, out, qs, tile, lane);
}

// ---------------------------------------------------------------------------
// Host launcher
// ---------------------------------------------------------------------------
extern "C" void kernel_launch(void* const* d_in, const int* in_sizes, int n_in,
                              void* d_out, int out_size, void* d_ws, size_t ws_size,
                              hipStream_t stream) {
    constexpr int L = 2, T = 512, B = 32, I = 512, H = 512;
    constexpr unsigned long long QS = (unsigned long long)L * T * B * H;

    const float* x    = (const float*)d_in[0];
    const float* wihf = (const float*)d_in[1];
    const float* whhf = (const float*)d_in[2];
    const float* bih  = (const float*)d_in[3];
    const float* bhh  = (const float*)d_in[4];
    float* out = (float*)d_out;

    // Workspace layout (bf16): x (16MB) | h_layer0 (16MB) | h_layer1 (16MB)
    //                          | Wih (4MB) | Whh (4MB)   total ~56MB
    bf16_t* xb   = (bf16_t*)d_ws;
    bf16_t* hb0  = xb   + (size_t)T * B * I;
    bf16_t* hb1  = hb0  + (size_t)T * B * H;
    bf16_t* wihb = hb1  + (size_t)T * B * H;
    bf16_t* whhb = wihb + (size_t)L * 4 * H * I;

    cvt_f32_bf16<<<2048, 256, 0, stream>>>(x,    xb,   (long long)T * B * I);
    cvt_f32_bf16<<<1024, 256, 0, stream>>>(wihf, wihb, (long long)L * 4 * H * I);
    cvt_f32_bf16<<<1024, 256, 0, stream>>>(whhf, whhb, (long long)L * 4 * H * H);

    auto makeArgs = [&](int l, int t) -> StepArgs {
        StepArgs a;
        const bf16_t* xin = (l == 0) ? xb : hb0;
        bf16_t* hbl       = (l == 0) ? hb0 : hb1;
        a.xrow   = xin + (size_t)t * B * I;
        a.hprev  = (t > 0) ? (hbl + (size_t)(t - 1) * B * H) : nullptr;
        a.wih    = wihb + (size_t)l * 4 * H * I;
        a.whh    = whhb + (size_t)l * 4 * H * H;
        a.bih    = bih + (size_t)l * 4 * H;
        a.bhh    = bhh + (size_t)l * 4 * H;
        a.cprev  = (t > 0)
                 ? (out + QS + ((unsigned long long)(l * T + t - 1) * B) * H)
                 : nullptr;
        a.hout   = hbl + (size_t)t * B * H;
        a.outOff = ((unsigned long long)(l * T + t) * B) * H;
        return a;
    };

    // Software pipeline across layers: at super-step s run layer0 step s and
    // layer1 step s-1 in one grid (layer1@t-1 only needs h0[t-1], written by
    // the previous launch). 513 step launches instead of 1024.
    for (int s = 0; s <= T; ++s) {
        StepArgs a0, a1;
        int njobs = 0;
        if (s < T) { a0 = makeArgs(0, s); ++njobs; }
        if (s >= 1) {
            StepArgs a = makeArgs(1, s - 1);
            if (njobs == 0) a0 = a; else a1 = a;
            ++njobs;
        }
        if (njobs == 1) a1 = a0;  // unused but keep defined
        lstm_step<<<njobs * 64, 32, 0, stream>>>(a0, a1, out, QS);
    }
}
